// WaveNet_39539468927086
// MI455X (gfx1250) — matrix operand order repaired and checked by reading, not compiled, not run on hardware
//
#include <hip/hip_runtime.h>
#include <hip/hip_bf16.h>
#include <cstdint>
#include <cstddef>

// CDNA5 / gfx1250 WaveNet forward. wave32; WMMA bf16 16x16x32, fp32 accum.
// Workspace layout (needs ~335MB): 3 x 64MB bf16 activation buffers (rotating),
// fp32 skip accumulator, bf16 weight copies, stats, mean accumulator.

typedef __attribute__((ext_vector_type(16))) __bf16 v16bf;
typedef __attribute__((ext_vector_type(8)))  float  v8f;

union AV { v16bf v; uint4 q[2]; };
union BV { v16bf v; uint4 q[2]; };

__device__ __forceinline__ unsigned short f32_to_bf16_rne(float f) {
  unsigned u = __builtin_bit_cast(unsigned, f);
  unsigned r = u + 0x7FFFu + ((u >> 16) & 1u);
  return (unsigned short)(r >> 16);
}
__device__ __forceinline__ float bf16_to_f32(unsigned short h) {
  unsigned u = ((unsigned)h) << 16;
  return __builtin_bit_cast(float, u);
}

#define LP 136   // LDS row pitch in ushorts (128 + 8 pad -> bank spread)

// ---------------- zero / convert helpers ----------------
__global__ __launch_bounds__(256) void k_zero(float* a, size_t na, float* b, size_t nb) {
  size_t i = (size_t)blockIdx.x * 256 + threadIdx.x;
  if (i < na) a[i] = 0.f;
  if (i < nb) b[i] = 0.f;
}

__global__ __launch_bounds__(256) void k_cvt(const float* __restrict__ s,
                                             unsigned short* __restrict__ d, int n) {
  int i = blockIdx.x * 256 + threadIdx.x;
  if (i < n) d[i] = f32_to_bf16_rne(s[i]);
}

// dil_w [L][C][C][2] fp32 -> [L][2][C][C] bf16   (128*128*2 == 1<<15)
__global__ __launch_bounds__(256) void k_cvt_dil(const float* __restrict__ s,
                                                 unsigned short* __restrict__ d, int n) {
  int j = blockIdx.x * 256 + threadIdx.x;
  if (j >= n) return;
  int k = j & 127, m = (j >> 7) & 127, tap = (j >> 14) & 1, i = j >> 15;
  d[j] = f32_to_bf16_rne(s[(((size_t)i * 128 + m) * 128 + k) * 2 + tap]);
}

// ---------------- init conv (2->128) + stats ----------------
__global__ __launch_bounds__(256) void k_init_conv(
    const float* __restrict__ x, const float* __restrict__ iw, const float* __restrict__ ib,
    unsigned short* __restrict__ pre, float* __restrict__ st, int T, int T0, int T0_s) {
  __shared__ float ls[16];
  const int b = blockIdx.y;
  const int t = blockIdx.x * 256 + threadIdx.x;
  if (threadIdx.x < 16) ls[threadIdx.x] = 0.f;
  __syncthreads();
  if (t < T0) {
    const float x0  = x[(size_t)b * 2 * T + t];
    const float x0n = x[(size_t)b * 2 * T + t + 1];
    const float x1  = x[((size_t)b * 2 + 1) * T + t];
    const float x1n = x[((size_t)b * 2 + 1) * T + t + 1];
    for (int g = 0; g < 8; ++g) {
      float g1 = 0.f, g2 = 0.f;
      for (int cc = 0; cc < 16; ++cc) {
        const int c = g * 16 + cc;
        const float y = iw[c*4+0]*x0 + iw[c*4+1]*x0n + iw[c*4+2]*x1 + iw[c*4+3]*x1n + ib[c];
        pre[((size_t)(b * 128 + c)) * T0_s + t] = f32_to_bf16_rne(y);
        g1 += y; g2 += y * y;
      }
      atomicAdd(&ls[g * 2], g1);
      atomicAdd(&ls[g * 2 + 1], g2);
    }
  }
  __syncthreads();
  if (threadIdx.x < 16) atomicAdd(&st[b * 16 + threadIdx.x], ls[threadIdx.x]);
}

// ---------------- GroupNorm in place (init layer) ----------------
__global__ __launch_bounds__(256) void k_gn_inplace(
    unsigned short* __restrict__ h, const float* __restrict__ st,
    const float* __restrict__ gg, const float* __restrict__ bb, int T0, int T0_s) {
  const int b = blockIdx.y;
  const int t = blockIdx.x * 256 + threadIdx.x;
  if (t >= T0) return;
  const float cnt = 16.f * (float)T0;
  for (int c = 0; c < 128; ++c) {
    const int g = c >> 4;
    const float mu  = st[(b * 8 + g) * 2] / cnt;
    const float var = fmaxf(st[(b * 8 + g) * 2 + 1] / cnt - mu * mu, 0.f);
    const float inv = rsqrtf(var + 1e-5f);
    const size_t idx = ((size_t)(b * 128 + c)) * T0_s + t;
    h[idx] = f32_to_bf16_rne((bf16_to_f32(h[idx]) - mu) * inv * gg[c] + bb[c]);
  }
}

// ---------------- K1: dilated conv (2-tap 128x128 WMMA) + stats ----------------
__global__ __launch_bounds__(256) void k_dilconv(
    const unsigned short* __restrict__ hin, const unsigned short* __restrict__ w2,
    const float* __restrict__ bias, unsigned short* __restrict__ yout,
    float* __restrict__ stats, int Tin, int Tin_s, int Tout, int Tout_s, int dil) {
  __shared__ unsigned short lds[(64 + 32) * LP];
  const int b = blockIdx.y;
  const int t0 = blockIdx.x * 64;
  const int tid = threadIdx.x, wave = tid >> 5, lane = tid & 31;
  const int tcols = 64 + dil;

  // Stage [t0, t0+64+dil) x 128ch, transposed [c][t] -> [t][c] in LDS (bf16).
  for (int cc = 0; cc < 16; ++cc) {
    const int c = wave * 16 + cc;
    const unsigned short* src = hin + ((size_t)(b * 128 + c)) * Tin_s + t0;
    __builtin_prefetch(src + tcols, 0, 1);  // next tile -> global_prefetch_b8
    for (int tt = lane * 2; tt < tcols; tt += 64) {
      if (t0 + tt < Tin) {
        const unsigned pair = *(const unsigned*)(src + tt);  // Tin_s even, tt even
        lds[tt * LP + c]       = (unsigned short)(pair & 0xFFFFu);
        lds[(tt + 1) * LP + c] = (unsigned short)(pair >> 16);
      }
    }
  }
  __syncthreads();

  // A tiles (weights, both taps) - 16x32 bf16 per K-chunk, contiguous k = 16*half + s per lane (B matches).
  AV a[2][4];
  {
    const int mrow = lane & 15, half = lane >> 4;
    for (int tap = 0; tap < 2; ++tap)
      for (int kc = 0; kc < 4; ++kc) {
        const unsigned short* p = w2 + ((size_t)tap * 128 + wave * 16 + mrow) * 128
                                  + kc * 32 + half * 16;
        a[tap][kc].q[0] = *(const uint4*)p;
        a[tap][kc].q[1] = *(const uint4*)(p + 8);
      }
  }

  float s1 = 0.f, s2 = 0.f;
  const int n = lane & 15, half = lane >> 4;
  for (int nt = 0; nt < 4; ++nt) {
    v8f acc = {0.f, 0.f, 0.f, 0.f, 0.f, 0.f, 0.f, 0.f};
    const int trow = nt * 16 + n;
    for (int kc = 0; kc < 4; ++kc) {
      BV b0, b1;
      const unsigned short* p0 = &lds[trow * LP + kc * 32 + half * 16];
      b0.q[0] = *(const uint4*)p0; b0.q[1] = *(const uint4*)(p0 + 8);
      const unsigned short* p1 = &lds[(trow + dil) * LP + kc * 32 + half * 16];
      b1.q[0] = *(const uint4*)p1; b1.q[1] = *(const uint4*)(p1 + 8);
      acc = __builtin_amdgcn_wmma_f32_16x16x32_bf16(false, a[0][kc].v, false, b0.v,
                                                    (short)0, acc, false, false);
      acc = __builtin_amdgcn_wmma_f32_16x16x32_bf16(false, a[1][kc].v, false, b1.v,
                                                    (short)0, acc, false, false);
    }
    const int tg = t0 + nt * 16 + n;
    if (tg < Tout) {
      for (int r = 0; r < 8; ++r) {
        const int mch = wave * 16 + r + 8 * half;
        const float y = acc[r] + bias[mch];
        yout[((size_t)(b * 128 + mch)) * Tout_s + tg] = f32_to_bf16_rne(y);
        s1 += y; s2 += y * y;
      }
    }
  }
  // whole wave belongs to group == wave (16 ch/group)
  for (int off = 16; off >= 1; off >>= 1) {
    s1 += __shfl_xor(s1, off, 32);
    s2 += __shfl_xor(s2, off, 32);
  }
  if (lane == 0) {
    atomicAdd(&stats[(b * 8 + wave) * 2 + 0], s1);
    atomicAdd(&stats[(b * 8 + wave) * 2 + 1], s2);
  }
}

// ---------------- K2: GN+gate fused into load, skip & res WMMA ----------------
__global__ __launch_bounds__(256) void k_gate(
    const unsigned short* __restrict__ tpre, const float* __restrict__ st1,
    const float* __restrict__ g1, const float* __restrict__ b1v,
    const unsigned short* __restrict__ wskip, const float* __restrict__ skip_b,
    const unsigned short* __restrict__ wres, const float* __restrict__ res_b,
    float* __restrict__ skip_sum, unsigned short* __restrict__ rpre,
    float* __restrict__ st2, int Tout, int Tout_s, int out_sz, int out_s, int has_res) {
  __shared__ unsigned short lds[64 * LP];
  const int b = blockIdx.y, t0 = blockIdx.x * 64;
  const int tid = threadIdx.x, wave = tid >> 5, lane = tid & 31;
  const float cnt = 16.f * (float)Tout;
  const float mu  = st1[(b * 8 + wave) * 2 + 0] / cnt;
  const float var = fmaxf(st1[(b * 8 + wave) * 2 + 1] / cnt - mu * mu, 0.f);
  const float inv = rsqrtf(var + 1e-5f);

  for (int cc = 0; cc < 16; ++cc) {
    const int c = wave * 16 + cc;
    const float gs = g1[c] * inv;
    const float bs = b1v[c] - mu * gs;
    const unsigned short* src = tpre + ((size_t)(b * 128 + c)) * Tout_s + t0;
    const int tt = lane * 2;
    if (t0 + tt < Tout) {
      const unsigned pair = *(const unsigned*)(src + tt);
      const float z0 = bf16_to_f32((unsigned short)(pair & 0xFFFFu)) * gs + bs;
      const float z1 = bf16_to_f32((unsigned short)(pair >> 16)) * gs + bs;
      lds[tt * LP + c]       = f32_to_bf16_rne(tanhf(z0) / (1.f + __expf(-z0)));
      lds[(tt + 1) * LP + c] = f32_to_bf16_rne(tanhf(z1) / (1.f + __expf(-z1)));
    }
  }
  __syncthreads();

  AV as[4], ar[4];
  {
    const int mrow = lane & 15, half = lane >> 4;
    for (int kc = 0; kc < 4; ++kc) {
      const unsigned short* ps = wskip + ((size_t)(wave * 16 + mrow)) * 128 + kc * 32 + half * 16;
      as[kc].q[0] = *(const uint4*)ps; as[kc].q[1] = *(const uint4*)(ps + 8);
      if (has_res) {
        const unsigned short* pr = wres + ((size_t)(wave * 16 + mrow)) * 128 + kc * 32 + half * 16;
        ar[kc].q[0] = *(const uint4*)pr; ar[kc].q[1] = *(const uint4*)(pr + 8);
      }
    }
  }

  const int skip_off = Tout - out_sz;
  float s1 = 0.f, s2 = 0.f;
  const int n = lane & 15, half = lane >> 4;
  for (int nt = 0; nt < 4; ++nt) {
    v8f accS = {0.f,0.f,0.f,0.f,0.f,0.f,0.f,0.f};
    v8f accR = {0.f,0.f,0.f,0.f,0.f,0.f,0.f,0.f};
    for (int kc = 0; kc < 4; ++kc) {
      BV bb;
      const unsigned short* p = &lds[(nt * 16 + n) * LP + kc * 32 + half * 16];
      bb.q[0] = *(const uint4*)p; bb.q[1] = *(const uint4*)(p + 8);
      accS = __builtin_amdgcn_wmma_f32_16x16x32_bf16(false, as[kc].v, false, bb.v,
                                                     (short)0, accS, false, false);
      if (has_res)
        accR = __builtin_amdgcn_wmma_f32_16x16x32_bf16(false, ar[kc].v, false, bb.v,
                                                       (short)0, accR, false, false);
    }
    const int tg = t0 + nt * 16 + n;
    if (tg < Tout) {
      for (int r = 0; r < 8; ++r) {
        const int mch = wave * 16 + r + 8 * half;
        const float sv = accS[r] + skip_b[mch];
        const int ts = tg - skip_off;
        if (ts >= 0) skip_sum[((size_t)(b * 128 + mch)) * out_s + ts] += sv;
        if (has_res) {
          const float yv = accR[r] + res_b[mch];
          rpre[((size_t)(b * 128 + mch)) * Tout_s + tg] = f32_to_bf16_rne(yv);
          s1 += yv; s2 += yv * yv;
        }
      }
    }
  }
  if (has_res) {
    for (int off = 16; off >= 1; off >>= 1) {
      s1 += __shfl_xor(s1, off, 32);
      s2 += __shfl_xor(s2, off, 32);
    }
    if (lane == 0) {
      atomicAdd(&st2[(b * 8 + wave) * 2 + 0], s1);
      atomicAdd(&st2[(b * 8 + wave) * 2 + 1], s2);
    }
  }
}

// ---------------- K3: GN(res) + residual add ----------------
__global__ __launch_bounds__(256) void k_resadd(
    const unsigned short* __restrict__ rpre, const float* __restrict__ st2,
    const float* __restrict__ g2, const float* __restrict__ b2v,
    const unsigned short* __restrict__ hold, unsigned short* __restrict__ hnew,
    int Tout, int Tout_s, int Tin_s, int dil) {
  const int b = blockIdx.y;
  const int t = blockIdx.x * 256 + threadIdx.x;
  if (t >= Tout) return;
  const float cnt = 16.f * (float)Tout;
  for (int c = 0; c < 128; ++c) {
    const int g = c >> 4;
    const float mu  = st2[(b * 8 + g) * 2] / cnt;
    const float var = fmaxf(st2[(b * 8 + g) * 2 + 1] / cnt - mu * mu, 0.f);
    const float inv = rsqrtf(var + 1e-5f);
    const float y = bf16_to_f32(rpre[((size_t)(b * 128 + c)) * Tout_s + t]);
    const float hn = (y - mu) * inv * g2[c] + b2v[c]
                   + bf16_to_f32(hold[((size_t)(b * 128 + c)) * Tin_s + t + dil]);
    hnew[((size_t)(b * 128 + c)) * Tout_s + t] = f32_to_bf16_rne(hn);
  }
}

// ---------------- head: relu -> d1 -> relu -> d2 -> column sums ----------------
__global__ __launch_bounds__(256) void k_head(
    const float* __restrict__ skip_sum, const unsigned short* __restrict__ wd1,
    const float* __restrict__ d1b, const unsigned short* __restrict__ wd2,
    const float* __restrict__ d2b, float* __restrict__ mean_acc, int out_sz, int out_s) {
  __shared__ unsigned short in_t[64 * LP];
  __shared__ unsigned short o1_t[64 * LP];
  const int b = blockIdx.y, t0 = blockIdx.x * 64;
  const int tid = threadIdx.x, wave = tid >> 5, lane = tid & 31;

  for (int cc = 0; cc < 16; ++cc) {
    const int c = wave * 16 + cc;
    const float* src = skip_sum + ((size_t)(b * 128 + c)) * out_s + t0;
    const int tt = lane * 2;
    if (t0 + tt < out_sz) {
      const float2 v = *(const float2*)(src + tt);
      in_t[tt * LP + c]       = f32_to_bf16_rne(fmaxf(v.x, 0.f));
      in_t[(tt + 1) * LP + c] = f32_to_bf16_rne(fmaxf(v.y, 0.f));
    }
  }
  __syncthreads();

  AV a1[4], a2[4];
  const int mrow = lane & 15, half = lane >> 4, n = lane & 15;
  for (int kc = 0; kc < 4; ++kc) {
    const unsigned short* p1 = wd1 + ((size_t)(wave * 16 + mrow)) * 128 + kc * 32 + half * 16;
    a1[kc].q[0] = *(const uint4*)p1; a1[kc].q[1] = *(const uint4*)(p1 + 8);
    const unsigned short* p2 = wd2 + ((size_t)(wave * 16 + mrow)) * 128 + kc * 32 + half * 16;
    a2[kc].q[0] = *(const uint4*)p2; a2[kc].q[1] = *(const uint4*)(p2 + 8);
  }

  for (int nt = 0; nt < 4; ++nt) {
    v8f acc = {0.f,0.f,0.f,0.f,0.f,0.f,0.f,0.f};
    for (int kc = 0; kc < 4; ++kc) {
      BV bb;
      const unsigned short* p = &in_t[(nt * 16 + n) * LP + kc * 32 + half * 16];
      bb.q[0] = *(const uint4*)p; bb.q[1] = *(const uint4*)(p + 8);
      acc = __builtin_amdgcn_wmma_f32_16x16x32_bf16(false, a1[kc].v, false, bb.v,
                                                    (short)0, acc, false, false);
    }
    for (int r = 0; r < 8; ++r) {
      const int mch = wave * 16 + r + 8 * half;
      o1_t[(nt * 16 + n) * LP + mch] = f32_to_bf16_rne(fmaxf(acc[r] + d1b[mch], 0.f));
    }
  }
  __syncthreads();

  float csum[8] = {0.f,0.f,0.f,0.f,0.f,0.f,0.f,0.f};
  for (int nt = 0; nt < 4; ++nt) {
    v8f acc = {0.f,0.f,0.f,0.f,0.f,0.f,0.f,0.f};
    for (int kc = 0; kc < 4; ++kc) {
      BV bb;
      const unsigned short* p = &o1_t[(nt * 16 + n) * LP + kc * 32 + half * 16];
      bb.q[0] = *(const uint4*)p; bb.q[1] = *(const uint4*)(p + 8);
      acc = __builtin_amdgcn_wmma_f32_16x16x32_bf16(false, a2[kc].v, false, bb.v,
                                                    (short)0, acc, false, false);
    }
    const int tg = t0 + nt * 16 + n;
    if (tg < out_sz)
      for (int r = 0; r < 8; ++r)
        csum[r] += acc[r] + d2b[wave * 16 + r + 8 * half];
  }
  for (int r = 0; r < 8; ++r) {
    float v = csum[r];
    for (int off = 8; off >= 1; off >>= 1) v += __shfl_xor(v, off, 32);
    if ((lane & 15) == 0)
      atomicAdd(&mean_acc[b * 128 + wave * 16 + r + 8 * half], v);
  }
}

__global__ void k_final(const float* __restrict__ mean_acc, const float* __restrict__ lin_w,
                        const float* __restrict__ lin_b, float* __restrict__ out, int out_sz) {
  const int b = threadIdx.x;
  if (b >= 32) return;
  float s = 0.f;
  for (int c = 0; c < 128; ++c) s += lin_w[c] * mean_acc[b * 128 + c];
  out[b] = s / (float)out_sz + lin_b[0];
}

// ---------------- host side ----------------
extern "C" void kernel_launch(void* const* d_in, const int* in_sizes, int n_in,
                              void* d_out, int out_size, void* d_ws, size_t ws_size,
                              hipStream_t stream) {
  (void)in_sizes; (void)n_in; (void)out_size; (void)ws_size;
  const float* x      = (const float*)d_in[0];
  const float* init_w = (const float*)d_in[1];
  const float* init_b = (const float*)d_in[2];
  const float* n0_g   = (const float*)d_in[3];
  const float* n0_b   = (const float*)d_in[4];
  const float* dil_w  = (const float*)d_in[5];
  const float* dil_b  = (const float*)d_in[6];
  const float* n1_g   = (const float*)d_in[7];
  const float* n1_b   = (const float*)d_in[8];
  const float* res_w  = (const float*)d_in[9];
  const float* res_b  = (const float*)d_in[10];
  const float* n2_g   = (const float*)d_in[11];
  const float* n2_b   = (const float*)d_in[12];
  const float* skip_w = (const float*)d_in[13];
  const float* skip_b = (const float*)d_in[14];
  const float* d1_w   = (const float*)d_in[15];
  const float* d1_b   = (const float*)d_in[16];
  const float* d2_w   = (const float*)d_in[17];
  const float* d2_b   = (const float*)d_in[18];
  const float* lin_w  = (const float*)d_in[19];
  const float* lin_b  = (const float*)d_in[20];
  float* out = (float*)d_out;

  constexpr int Bn = 32, Cn = 128, Tn = 8192, Ln = 24;
  const int dils[Ln] = {1,2,4,8,16,32,1,2,4,8,16,32,1,2,4,8,16,32,1,2,4,8,16,32};
  const int T0 = Tn - 1;
  int rf = 0; for (int i = 0; i < Ln; ++i) rf += dils[i];
  const int out_sz = T0 - rf;   // 7939

  auto S = [](int t) { return (t + 8) & ~7; };  // padded time stride (even, >= t+1)
  const int S0 = S(T0);
  const int out_s = S(out_sz);

  char* ws = (char*)d_ws;
  const size_t BUF = (size_t)Bn * Cn * S0 * 2;         // 64MB bf16 activation buffer
  unsigned short* buf[3] = { (unsigned short*)ws,
                             (unsigned short*)(ws + BUF),
                             (unsigned short*)(ws + 2 * BUF) };
  char* p = ws + 3 * BUF;
  auto alloc = [&](size_t bytes) { char* r = p; p += (bytes + 255) & ~(size_t)255; return r; };

  const size_t SKIPN = (size_t)Bn * Cn * out_s;
  float* skip_sum = (float*)alloc(SKIPN * 4);
  unsigned short* wdil  = (unsigned short*)alloc((size_t)Ln * 2 * Cn * Cn * 2);
  unsigned short* wres  = (unsigned short*)alloc((size_t)(Ln - 1) * Cn * Cn * 2);
  unsigned short* wskip = (unsigned short*)alloc((size_t)Ln * Cn * Cn * 2);
  unsigned short* wd1   = (unsigned short*)alloc((size_t)Cn * Cn * 2);
  unsigned short* wd2   = (unsigned short*)alloc((size_t)Cn * Cn * 2);
  float* stats_all = (float*)alloc((48 * 512 + Bn * Cn) * 4);
  float* st_init = stats_all;               // [B][8][2]
  float* st1 = stats_all + 512;             // 24 x [B][8][2]
  float* st2 = stats_all + 512 * 25;        // 23 x [B][8][2]
  float* mean_acc = stats_all + 512 * 48;   // [B][128]

  // zero skip accumulator + stats + mean_acc
  const size_t nmisc = 48 * 512 + (size_t)Bn * Cn;
  k_zero<<<(unsigned)((SKIPN + 255) / 256), 256, 0, stream>>>(skip_sum, SKIPN, stats_all, nmisc);

  // convert weights to bf16 (deterministic; tiny)
  k_cvt_dil<<<(Ln * 2 * Cn * Cn + 255) / 256, 256, 0, stream>>>(dil_w, wdil, Ln * 2 * Cn * Cn);
  k_cvt<<<((Ln - 1) * Cn * Cn + 255) / 256, 256, 0, stream>>>(res_w, wres, (Ln - 1) * Cn * Cn);
  k_cvt<<<(Ln * Cn * Cn + 255) / 256, 256, 0, stream>>>(skip_w, wskip, Ln * Cn * Cn);
  k_cvt<<<(Cn * Cn + 255) / 256, 256, 0, stream>>>(d1_w, wd1, Cn * Cn);
  k_cvt<<<(Cn * Cn + 255) / 256, 256, 0, stream>>>(d2_w, wd2, Cn * Cn);

  // init conv + GN
  dim3 gi((T0 + 255) / 256, Bn);
  k_init_conv<<<gi, 256, 0, stream>>>(x, init_w, init_b, buf[0], st_init, Tn, T0, S0);
  k_gn_inplace<<<gi, 256, 0, stream>>>(buf[0], st_init, n0_g, n0_b, T0, S0);

  int cur = 0, Tin = T0;
  for (int i = 0; i < Ln; ++i) {
    const int d = dils[i];
    const int Tout = Tin - d;
    const int tp = (cur + 1) % 3, rp = (cur + 2) % 3;
    const int has_res = (i < Ln - 1) ? 1 : 0;
    dim3 g1((Tout + 63) / 64, Bn);

    k_dilconv<<<g1, 256, 0, stream>>>(buf[cur], wdil + (size_t)i * 2 * Cn * Cn,
                                      dil_b + i * Cn, buf[tp], st1 + i * 512,
                                      Tin, S(Tin), Tout, S(Tout), d);

    k_gate<<<g1, 256, 0, stream>>>(buf[tp], st1 + i * 512, n1_g + i * Cn, n1_b + i * Cn,
                                   wskip + (size_t)i * Cn * Cn, skip_b + i * Cn,
                                   has_res ? wres + (size_t)i * Cn * Cn : wres,
                                   has_res ? res_b + i * Cn : res_b,
                                   skip_sum, buf[rp],
                                   has_res ? st2 + i * 512 : st2,
                                   Tout, S(Tout), out_sz, out_s, has_res);

    if (has_res) {
      dim3 g3((Tout + 255) / 256, Bn);
      k_resadd<<<g3, 256, 0, stream>>>(buf[rp], st2 + i * 512, n2_g + i * Cn, n2_b + i * Cn,
                                       buf[cur], buf[tp], Tout, S(Tout), S(Tin), d);
      cur = tp;  // new h lives in old t_pre buffer
    }
    Tin = Tout;
  }

  dim3 gh((out_sz + 63) / 64, Bn);
  k_head<<<gh, 256, 0, stream>>>(skip_sum, wd1, d1_b, wd2, d2_b, mean_acc, out_sz, out_s);
  k_final<<<1, 64, 0, stream>>>(mean_acc, lin_w, lin_b, out, out_sz);
}
